// NES_15255723836206
// MI455X (gfx1250) — matrix operand order repaired and checked by reading, not compiled, not run on hardware
//
#include <hip/hip_runtime.h>
#include <hip/hip_bf16.h>

// ---------------------------------------------------------------------------
// Fused SlotAttention for MI455X (gfx1250), wave32 + WMMA bf16.
// One workgroup = 16 batch elements (48 rows). Everything stays in LDS
// (~293 KB of the 320 KB/WGP) across all 3 iterations; weights are streamed
// from L2 as bf16 (converted once per launch into d_ws by a prologue kernel).
// ---------------------------------------------------------------------------

typedef __bf16 bf16_t;
typedef __attribute__((ext_vector_type(16))) __bf16 v16bf;
typedef __attribute__((ext_vector_type(8)))  __bf16 v8bf;
typedef __attribute__((ext_vector_type(8)))  float  v8f;

#define DD    512
#define TT    3
#define SSL   3
#define NBAT  32768
#define PROJN 768
#define BBLK  16                 // batch elems per block
#define ROWS  48                 // BBLK * SSL (== BBLK * TT)
#define LDA   520                // padded bf16 row stride (bank-conflict free)
#define LDSF  516                // padded f32 row stride
#define SMEM_BYTES (4*ROWS*LDA*2 + ROWS*LDSF*4 + 2*144*4)   // 299904 B

static __device__ __forceinline__ v16bf cat8(v8bf lo, v8bf hi){
  return __builtin_shufflevector(lo, hi, 0,1,2,3,4,5,6,7,8,9,10,11,12,13,14,15);
}

// A fragment: 16x32 bf16 tile from LDS (row-major, stride LDA).
// lane<16: row=lane, K = kb..kb+7 then kb+16..kb+23 ; lane>=16: K shifted by 8.
static __device__ __forceinline__ v16bf fragA(const bf16_t* X, int mBase, int kb, int lane){
  int r = mBase + (lane & 15);
  int k = kb + ((lane >> 4) << 3);
  const bf16_t* p = X + r * LDA + k;
  return cat8(*(const v8bf*)p, *(const v8bf*)(p + 16));
}

// B fragment: 32x16 tile of W^T where W is [N][512] row-major bf16 in global.
// lane holds K = kb..kb+7 then kb+16..kb+23 (lane>=16: shifted by 8) of output column n = nBase + (lane&15).
static __device__ __forceinline__ v16bf fragB(const bf16_t* W, int nBase, int kb, int lane){
  int n = nBase + (lane & 15);
  int k = kb + ((lane >> 4) << 3);
  const bf16_t* p = W + n * DD + k;
  return cat8(*(const v8bf*)p, *(const v8bf*)(p + 16));
}

static __device__ __forceinline__ v8f wmma_bf(v16bf a, v16bf b, v8f c){
  return __builtin_amdgcn_wmma_f32_16x16x32_bf16(false, a, false, b, (short)0, c, false, false);
}

static __device__ __forceinline__ v8f splat8(float x){
  v8f v;
  #pragma unroll
  for (int i = 0; i < 8; ++i) v[i] = x;
  return v;
}

static __device__ __forceinline__ v8f gemm512(const bf16_t* X, const bf16_t* W,
                                              int mBase, int nBase, v8f c, int lane){
  #pragma unroll 4
  for (int kt = 0; kt < 16; ++kt){
    int kb = kt * 32;
    v16bf a = fragA(X, mBase, kb, lane);
    v16bf b = fragB(W, nBase, kb, lane);
    c = wmma_bf(a, b, c);
  }
  return c;
}

// C tile store (f32 acc -> bf16 LDS). VGPR r -> row mBase + r + 8*(lane>=16).
static __device__ __forceinline__ void store_bf(bf16_t* dst, v8f c, int mBase, int nBase, int lane){
  int n  = nBase + (lane & 15);
  int mo = mBase + ((lane >> 4) << 3);
  #pragma unroll
  for (int r = 0; r < 8; ++r) dst[(mo + r) * LDA + n] = (bf16_t)c[r];
}

static __device__ __forceinline__ float wred(float x){
  #pragma unroll
  for (int m = 16; m >= 1; m >>= 1) x += __shfl_xor(x, m, 32);
  return x;
}

static __device__ __forceinline__ float sigmf(float x){ return 1.0f / (1.0f + __expf(-x)); }

static __device__ __forceinline__ unsigned pcg(unsigned x){
  x = x * 747796405u + 2891336453u;
  unsigned w = ((x >> ((x >> 28u) + 4u)) ^ x) * 277803737u;
  return (w >> 22u) ^ w;
}

// LayerNorm of 48 LDS f32 rows -> bf16 rows (with affine). 6 rows per wave.
static __device__ __forceinline__ void ln_rows(const float* S, bf16_t* dst,
                                               const float* g, const float* b,
                                               int wave, int lane){
  int d0 = lane * 16;
  for (int r = wave * 6; r < wave * 6 + 6; ++r){
    const float* row = S + r * LDSF;
    float x[16]; float s = 0.f, ss = 0.f;
    #pragma unroll
    for (int e = 0; e < 16; ++e){ float v = row[d0 + e]; x[e] = v; s += v; ss += v * v; }
    s = wred(s); ss = wred(ss);
    float mean = s * (1.0f / 512.0f);
    float var  = ss * (1.0f / 512.0f) - mean * mean;
    float inv  = rsqrtf(var + 1e-5f);
    bf16_t* dr = dst + r * LDA;
    #pragma unroll
    for (int e = 0; e < 16; ++e){
      int d = d0 + e;
      dr[d] = (bf16_t)(((x[e] - mean) * inv) * g[d] + b[d]);
    }
  }
}

static __device__ __forceinline__ void f32_to_bf_rows(const float* S, bf16_t* dst,
                                                      int wave, int lane){
  int d0 = lane * 16;
  for (int r = wave * 6; r < wave * 6 + 6; ++r){
    const float* row = S + r * LDSF;
    bf16_t* dr = dst + r * LDA;
    #pragma unroll
    for (int e = 0; e < 16; ++e) dr[d0 + e] = (bf16_t)row[d0 + e];
  }
}

// ---------------------------------------------------------------------------

__global__ void NES_cvt_bf16(const float* __restrict__ src, bf16_t* __restrict__ dst, int n){
  int i = blockIdx.x * blockDim.x + threadIdx.x;
  if (i < n) dst[i] = (bf16_t)src[i];
}

__global__ __launch_bounds__(256, 1)
void NES_slot_attn(const int* __restrict__ tokens, const unsigned char* __restrict__ mask,
                   const float* __restrict__ tok_emb, const float* __restrict__ pad_emb,
                   const float* __restrict__ mu, const float* __restrict__ lsig,
                   const float* __restrict__ bq, const float* __restrict__ bk,
                   const float* __restrict__ bv,
                   const float* __restrict__ bih, const float* __restrict__ bhh,
                   const float* __restrict__ b1, const float* __restrict__ b2,
                   const float* __restrict__ ln_in_g, const float* __restrict__ ln_in_b,
                   const float* __restrict__ ln_sl_g, const float* __restrict__ ln_sl_b,
                   const float* __restrict__ ln_ff_g, const float* __restrict__ ln_ff_b,
                   const float* __restrict__ proj_bias,
                   const bf16_t* __restrict__ wqb, const bf16_t* __restrict__ wkb,
                   const bf16_t* __restrict__ wvb, const bf16_t* __restrict__ wihb,
                   const bf16_t* __restrict__ whhb, const bf16_t* __restrict__ w1b,
                   const bf16_t* __restrict__ w2b, const bf16_t* __restrict__ projb,
                   float* __restrict__ out)
{
  extern __shared__ char smem[];
  bf16_t* bufA  = (bf16_t*)smem;            // generic A operand (xin / LN(slots) / updates / hidden)
  bf16_t* bufA2 = bufA  + ROWS * LDA;       // q / slots_prev / hidden
  bf16_t* bufK  = bufA2 + ROWS * LDA;       // k (persists all iterations)
  bf16_t* bufV  = bufK  + ROWS * LDA;       // v (persists)
  float*  bufS  = (float*)(bufV + ROWS * LDA);  // slots, f32, [48][LDSF]
  float*  sDots = bufS + ROWS * LDSF;       // [16][3][3]
  float*  sAttn = sDots + 144;

  const int tid  = threadIdx.x;
  const int wave = tid >> 5;
  const int lane = tid & 31;
  const int d0   = lane * 16;

  // ---- phase 0: x = where(mask, emb+posenc, pad); LN_in -> bufA (bf16) ----
  for (int r = wave * 6; r < wave * 6 + 6; ++r){
    int bb = r / 3, t = r % 3;
    int bg = blockIdx.x * BBLK + bb;
    int tok  = tokens[bg * TT + t];
    bool mk  = mask[bg * TT + t] != 0;
    const float* er = tok_emb + tok * DD;
    const float* pr = pad_emb + t * DD;
    float x[16]; float s = 0.f, ss = 0.f;
    #pragma unroll
    for (int e = 0; e < 16; ++e){
      int d = d0 + e;
      float val;
      if (mk){
        float invf = __expf(-(float)(d & ~1) * (9.210340372f / 512.0f)); // 10000^(-2i/D)
        float arg  = (float)t * invf;
        val = er[d] + ((d & 1) ? __cosf(arg) : __sinf(arg));
      } else {
        val = pr[d];
      }
      x[e] = val; s += val; ss += val * val;
    }
    s = wred(s); ss = wred(ss);
    float mean = s * (1.0f / 512.0f);
    float var  = ss * (1.0f / 512.0f) - mean * mean;
    float inv  = rsqrtf(var + 1e-5f);
    bf16_t* dr = bufA + r * LDA;
    #pragma unroll
    for (int e = 0; e < 16; ++e){
      int d = d0 + e;
      dr[d] = (bf16_t)(((x[e] - mean) * inv) * ln_in_g[d] + ln_in_b[d]);
    }
  }

  // ---- slots init: mu + exp(logsigma) * N(0,1)  (hash-based Box-Muller) ----
  for (int r = wave * 6; r < wave * 6 + 6; ++r){
    int bg = blockIdx.x * BBLK + r / 3;
    int sl = r % 3;
    float* dr = bufS + r * LDSF;
    #pragma unroll
    for (int e = 0; e < 16; ++e){
      int d = d0 + e;
      unsigned key = ((unsigned)(bg * SSL + sl) << 10) | (unsigned)d;
      unsigned h1 = pcg(key * 2u + 1u), h2 = pcg(key * 2u + 2u);
      float u1 = ((float)h1 + 1.0f) * 2.3283064365386963e-10f;
      float u2 = (float)h2 * 2.3283064365386963e-10f;
      float z  = sqrtf(-2.0f * __logf(u1)) * __cosf(6.283185307f * u2);
      dr[d] = mu[d] + __expf(lsig[d]) * z;
    }
  }
  __syncthreads();

  // ---- k = xin@wk.T + bk ; v = xin@wv.T + bv  (shared A fragments) ----
  for (int t = wave; t < 96; t += 8){
    int mT = t % 3, nT = t / 3;
    int mBase = mT * 16, nBase = nT * 16;
    int n = nBase + (lane & 15);
    v8f ck = splat8(bk[n]);
    v8f cv = splat8(bv[n]);
    #pragma unroll 2
    for (int kt = 0; kt < 16; ++kt){
      int kb = kt * 32;
      v16bf a = fragA(bufA, mBase, kb, lane);
      ck = wmma_bf(a, fragB(wkb, nBase, kb, lane), ck);
      cv = wmma_bf(a, fragB(wvb, nBase, kb, lane), cv);
    }
    store_bf(bufK, ck, mBase, nBase, lane);
    store_bf(bufV, cv, mBase, nBase, lane);
  }
  __syncthreads();

  // ======================= slot-attention iterations =======================
  for (int it = 0; it < 3; ++it){
    // LN_slots -> bufA
    ln_rows(bufS, bufA, ln_sl_g, ln_sl_b, wave, lane);
    __syncthreads();

    // q = LN(slots)@wq.T + bq -> bufA2
    for (int t = wave; t < 96; t += 8){
      int mT = t % 3, nT = t / 3;
      int mBase = mT * 16, nBase = nT * 16;
      int n = nBase + (lane & 15);
      v8f c = splat8(bq[n]);
      c = gemm512(bufA, wqb, mBase, nBase, c, lane);
      store_bf(bufA2, c, mBase, nBase, lane);
    }
    __syncthreads();

    // dots[b,i,j] = scale * q_i . k_j   (wave-cooperative dot products)
    for (int bb = wave; bb < BBLK; bb += 8){
      for (int p = 0; p < 9; ++p){
        int i = p / 3, j = p % 3;
        const bf16_t* qr = bufA2 + (bb * 3 + i) * LDA;
        const bf16_t* kr = bufK  + (bb * 3 + j) * LDA;
        float s = 0.f;
        #pragma unroll
        for (int e = 0; e < 16; ++e){ int d = d0 + e; s += (float)qr[d] * (float)kr[d]; }
        s = wred(s);
        if (lane == 0) sDots[bb * 9 + i * 3 + j] = s * 0.044194173824159216f; // 512^-0.5
      }
    }
    __syncthreads();

    // softmax over slots (axis i) per (b, j), + eps
    if (tid < 48){
      int bb = tid / 3, j = tid % 3;
      float v0 = sDots[bb*9 + 0 + j], v1 = sDots[bb*9 + 3 + j], v2 = sDots[bb*9 + 6 + j];
      float mx = fmaxf(v0, fmaxf(v1, v2));
      float e0 = __expf(v0 - mx), e1 = __expf(v1 - mx), e2 = __expf(v2 - mx);
      float inv = 1.0f / (e0 + e1 + e2);
      sAttn[bb*9 + 0 + j] = e0 * inv + 1e-8f;
      sAttn[bb*9 + 3 + j] = e1 * inv + 1e-8f;
      sAttn[bb*9 + 6 + j] = e2 * inv + 1e-8f;
    }
    __syncthreads();
    // normalize over inputs (axis j) per (b, i)
    if (tid < 48){
      int bb = tid / 3, i = tid % 3;
      float a0 = sAttn[bb*9 + i*3 + 0], a1 = sAttn[bb*9 + i*3 + 1], a2 = sAttn[bb*9 + i*3 + 2];
      float inv = 1.0f / (a0 + a1 + a2);
      sAttn[bb*9 + i*3 + 0] = a0 * inv;
      sAttn[bb*9 + i*3 + 1] = a1 * inv;
      sAttn[bb*9 + i*3 + 2] = a2 * inv;
    }
    __syncthreads();

    // updates = attn @ v -> bufA (bf16)
    for (int r = wave * 6; r < wave * 6 + 6; ++r){
      int bb = r / 3, i = r % 3;
      float a0 = sAttn[bb*9 + i*3 + 0], a1 = sAttn[bb*9 + i*3 + 1], a2 = sAttn[bb*9 + i*3 + 2];
      const bf16_t* v0 = bufV + (bb*3 + 0) * LDA;
      const bf16_t* v1 = bufV + (bb*3 + 1) * LDA;
      const bf16_t* v2 = bufV + (bb*3 + 2) * LDA;
      bf16_t* dr = bufA + r * LDA;
      #pragma unroll
      for (int e = 0; e < 16; ++e){
        int d = d0 + e;
        dr[d] = (bf16_t)(a0 * (float)v0[d] + a1 * (float)v1[d] + a2 * (float)v2[d]);
      }
    }
    // slots_prev (bf16) -> bufA2
    f32_to_bf_rows(bufS, bufA2, wave, lane);
    __syncthreads();

    // GRU: 6 fused gate GEMM tiles, gates in registers, h written to bufS
    for (int t = wave; t < 96; t += 8){
      int mT = t % 3, nT = t / 3;
      int mBase = mT * 16, nBase = nT * 16;
      int n = nBase + (lane & 15);
      v8f cir = splat8(bih[n]),        ciz = splat8(bih[512 + n]), cin_ = splat8(bih[1024 + n]);
      v8f chr_= splat8(bhh[n]),        chz = splat8(bhh[512 + n]), chn  = splat8(bhh[1024 + n]);
      #pragma unroll 2
      for (int kt = 0; kt < 16; ++kt){
        int kb = kt * 32;
        v16bf aU = fragA(bufA,  mBase, kb, lane);   // updates
        v16bf aS = fragA(bufA2, mBase, kb, lane);   // slots_prev
        cir  = wmma_bf(aU, fragB(wihb, nBase,        kb, lane), cir);
        ciz  = wmma_bf(aU, fragB(wihb, nBase + 512,  kb, lane), ciz);
        cin_ = wmma_bf(aU, fragB(wihb, nBase + 1024, kb, lane), cin_);
        chr_ = wmma_bf(aS, fragB(whhb, nBase,        kb, lane), chr_);
        chz  = wmma_bf(aS, fragB(whhb, nBase + 512,  kb, lane), chz);
        chn  = wmma_bf(aS, fragB(whhb, nBase + 1024, kb, lane), chn);
      }
      int mo = mBase + ((lane >> 4) << 3);
      #pragma unroll
      for (int r = 0; r < 8; ++r){
        int m = mo + r;
        float rg = sigmf(cir[r] + chr_[r]);
        float zg = sigmf(ciz[r] + chz[r]);
        float nn = tanhf(cin_[r] + rg * chn[r]);
        float hp = bufS[m * LDSF + n];
        bufS[m * LDSF + n] = (1.0f - zg) * nn + zg * hp;
      }
    }
    __syncthreads();

    // MLP: slots += relu(LN_ff(slots)@w1.T + b1) @ w2.T + b2
    ln_rows(bufS, bufA, ln_ff_g, ln_ff_b, wave, lane);
    __syncthreads();
    for (int t = wave; t < 96; t += 8){
      int mT = t % 3, nT = t / 3;
      int mBase = mT * 16, nBase = nT * 16;
      int n = nBase + (lane & 15);
      v8f c = splat8(b1[n]);
      c = gemm512(bufA, w1b, mBase, nBase, c, lane);
      #pragma unroll
      for (int r = 0; r < 8; ++r) c[r] = fmaxf(c[r], 0.0f);   // ReLU
      store_bf(bufA2, c, mBase, nBase, lane);
    }
    __syncthreads();
    for (int t = wave; t < 96; t += 8){
      int mT = t % 3, nT = t / 3;
      int mBase = mT * 16, nBase = nT * 16;
      int n  = nBase + (lane & 15);
      int mo = mBase + ((lane >> 4) << 3);
      v8f c;
      #pragma unroll
      for (int r = 0; r < 8; ++r) c[r] = b2[n] + bufS[(mo + r) * LDSF + n]; // residual init
      c = gemm512(bufA2, w2b, mBase, nBase, c, lane);
      #pragma unroll
      for (int r = 0; r < 8; ++r) bufS[(mo + r) * LDSF + n] = c[r];
    }
    __syncthreads();
  }

  // ---- final: xf_proj = slots@proj_w.T + proj_b ; also emit slots ----
  f32_to_bf_rows(bufS, bufA, wave, lane);
  __syncthreads();

  for (int t = wave; t < 144; t += 8){            // 3 mT x 48 nT (N = 768)
    int mT = t % 3, nT = t / 3;
    int mBase = mT * 16, nBase = nT * 16;
    int n = nBase + (lane & 15);
    v8f c = splat8(proj_bias[n]);
    c = gemm512(bufA, projb, mBase, nBase, c, lane);
    int mo = mBase + ((lane >> 4) << 3);
    #pragma unroll
    for (int r = 0; r < 8; ++r){
      int R = blockIdx.x * ROWS + mo + r;         // global (b*S + s) row
      out[(size_t)R * PROJN + n] = c[r];
    }
  }
  // slots output region starts after xf_proj: B*S*PROJ floats
  const size_t slotOff = (size_t)NBAT * SSL * PROJN;
  for (int idx = tid; idx < ROWS * DD; idx += 256){
    int m = idx >> 9, d = idx & 511;
    int R = blockIdx.x * ROWS + m;
    out[slotOff + (size_t)R * DD + d] = bufS[m * LDSF + d];
  }
}

// ---------------------------------------------------------------------------

extern "C" void kernel_launch(void* const* d_in, const int* in_sizes, int n_in,
                              void* d_out, int out_size, void* d_ws, size_t ws_size,
                              hipStream_t stream) {
  (void)in_sizes; (void)n_in; (void)out_size; (void)ws_size;

  const int*           tokens   = (const int*)d_in[0];
  const unsigned char* mask     = (const unsigned char*)d_in[1];
  const float* tok_emb = (const float*)d_in[2];
  const float* pad_emb = (const float*)d_in[3];
  const float* mu      = (const float*)d_in[4];
  const float* lsig    = (const float*)d_in[5];
  const float* wq      = (const float*)d_in[6];
  const float* bq      = (const float*)d_in[7];
  const float* wk      = (const float*)d_in[8];
  const float* bk      = (const float*)d_in[9];
  const float* wv      = (const float*)d_in[10];
  const float* bv      = (const float*)d_in[11];
  const float* gru_wih = (const float*)d_in[12];
  const float* gru_whh = (const float*)d_in[13];
  const float* gru_bih = (const float*)d_in[14];
  const float* gru_bhh = (const float*)d_in[15];
  const float* mlp_w1  = (const float*)d_in[16];
  const float* mlp_b1  = (const float*)d_in[17];
  const float* mlp_w2  = (const float*)d_in[18];
  const float* mlp_b2  = (const float*)d_in[19];
  const float* ln_in_g = (const float*)d_in[20];
  const float* ln_in_b = (const float*)d_in[21];
  const float* ln_sl_g = (const float*)d_in[22];
  const float* ln_sl_b = (const float*)d_in[23];
  const float* ln_ff_g = (const float*)d_in[24];
  const float* ln_ff_b = (const float*)d_in[25];
  const float* proj_w  = (const float*)d_in[26];
  const float* proj_b  = (const float*)d_in[27];

  // bf16 weight cache in workspace
  bf16_t* ws   = (bf16_t*)d_ws;
  bf16_t* wqb  = ws;
  bf16_t* wkb  = ws + 262144;
  bf16_t* wvb  = ws + 524288;
  bf16_t* wihb = ws + 786432;
  bf16_t* whhb = ws + 1572864;
  bf16_t* w1b  = ws + 2359296;
  bf16_t* w2b  = ws + 2621440;
  bf16_t* pjb  = ws + 2883584;   // end: 3276800 bf16 = 6.25 MB

  auto cvt = [&](const float* s, bf16_t* d, int n){
    NES_cvt_bf16<<<(n + 255) / 256, 256, 0, stream>>>(s, d, n);
  };
  cvt(wq,      wqb,  262144);
  cvt(wk,      wkb,  262144);
  cvt(wv,      wvb,  262144);
  cvt(gru_wih, wihb, 786432);
  cvt(gru_whh, whhb, 786432);
  cvt(mlp_w1,  w1b,  262144);
  cvt(mlp_w2,  w2b,  262144);
  cvt(proj_w,  pjb,  393216);

  NES_slot_attn<<<NBAT / BBLK, 256, SMEM_BYTES, stream>>>(
      tokens, mask, tok_emb, pad_emb, mu, lsig,
      bq, bk, bv, gru_bih, gru_bhh, mlp_b1, mlp_b2,
      ln_in_g, ln_in_b, ln_sl_g, ln_sl_b, ln_ff_g, ln_ff_b, proj_b,
      wqb, wkb, wvb, wihb, whhb, w1b, w2b, pjb,
      (float*)d_out);
}
